// ner_unit_24781961298183
// MI455X (gfx1250) — compile-verified, matrix operand order checked by reading, not run on hardware
//
#include <hip/hip_runtime.h>
#include <hip/hip_bf16.h>
#include <math.h>

typedef __attribute__((ext_vector_type(16))) _Float16 v16h;
typedef __attribute__((ext_vector_type(8)))  _Float16 v8h;
typedef __attribute__((ext_vector_type(8)))  float    v8f;

#define LDIM 96
#define BDIM 8
#define HDIM 768
#define TDIM 7

// -------- prep: fp32 -> f16 straight convert (for activations) --------
__global__ __launch_bounds__(256) void cvt_f16_kernel(const float* __restrict__ in,
                                                      _Float16* __restrict__ out, int n)
{
    int idx = blockIdx.x * blockDim.x + threadIdx.x;
    if (idx < n) out[idx] = (_Float16)in[idx];
}

// -------- prep: 768x768 fp32 -> f16 transpose (weights -> N-major) --------
// out[n*768 + k] = (f16) in[k*768 + n]
__global__ __launch_bounds__(256) void transpose_f16_kernel(const float* __restrict__ in,
                                                            _Float16* __restrict__ out)
{
    __shared__ float tile[32][33];
    int bk = blockIdx.x * 32;          // k base
    int bn = blockIdx.y * 32;          // n base
    int tx = threadIdx.x & 31;
    int ty = threadIdx.x >> 5;         // 0..7
    #pragma unroll
    for (int r = 0; r < 32; r += 8)
        tile[ty + r][tx] = in[(bk + ty + r) * HDIM + bn + tx];
    __syncthreads();
    #pragma unroll
    for (int r = 0; r < 32; r += 8)
        out[(bn + ty + r) * HDIM + bk + tx] = (_Float16)tile[tx][ty + r];
}

// Per CDNA5 16-bit fragment layout, each lane's v16h is two contiguous
// 16-byte chunks of its row: K = k0+hf*8..+7 and K = k0+16+hf*8..+7.
__device__ __forceinline__ v16h load_frag(const _Float16* __restrict__ row, int k0, int hf)
{
    v8h lo = *(const v8h*)(row + k0 + hf * 8);
    v8h hi = *(const v8h*)(row + k0 + 16 + hf * 8);
    v16h f;
    #pragma unroll
    for (int e = 0; e < 8; ++e) { f[e] = lo[e]; f[e + 8] = hi[e]; }
    return f;
}

// One wave per 16x64 output tile, 4 accumulators, A reused across 4 B tiles.
// blockIdx.z == 0 : G  = tanh(Xs@Wn1 + Xn@Wn2 + b_n)
// blockIdx.z == 1 : ST = Xn @ W1
// blockIdx.z == 2 : EN = Xn @ W2
__global__ __launch_bounds__(32) void gemm768_wmma_kernel(
    const _Float16* __restrict__ Xs, const _Float16* __restrict__ Xn,
    const _Float16* __restrict__ Wn1t, const _Float16* __restrict__ Wn2t,
    const _Float16* __restrict__ W1t,  const _Float16* __restrict__ W2t,
    const float* __restrict__ b_n,
    float* __restrict__ G, float* __restrict__ ST, float* __restrict__ EN)
{
    const int tm    = blockIdx.x * 16;
    const int tn    = blockIdx.y * 64;
    const int which = blockIdx.z;
    const int lane  = threadIdx.x;
    const int hf    = lane >> 4;       // half-wave selector
    const int r     = lane & 15;       // row (A) / col (B,C)

    const _Float16* Amat;  const _Float16* Bmat;
    const _Float16* Amat2; const _Float16* Bmat2;
    float* Out;
    if (which == 0)      { Amat = Xs; Bmat = Wn1t; Amat2 = Xn; Bmat2 = Wn2t; Out = G;  }
    else if (which == 1) { Amat = Xn; Bmat = W1t;  Amat2 = Xn; Bmat2 = W1t;  Out = ST; }
    else                 { Amat = Xn; Bmat = W2t;  Amat2 = Xn; Bmat2 = W2t;  Out = EN; }

    const _Float16* __restrict__ arow  = Amat  + (tm + r) * HDIM;
    const _Float16* __restrict__ arow2 = Amat2 + (tm + r) * HDIM;
    const _Float16* __restrict__ brow0 = Bmat  + (tn + r) * HDIM;
    const _Float16* __restrict__ brow20= Bmat2 + (tn + r) * HDIM;

    v8f acc[4] = {};
    for (int k0 = 0; k0 < HDIM; k0 += 32) {
        __builtin_prefetch(arow + k0 + 128, 0, 1);      // global_prefetch_b8
        __builtin_prefetch(brow0 + k0 + 128, 0, 1);
        v16h a = load_frag(arow, k0, hf);
        #pragma unroll
        for (int t = 0; t < 4; ++t) {
            v16h b = load_frag(brow0 + t * 16 * HDIM, k0, hf);
            acc[t] = __builtin_amdgcn_wmma_f32_16x16x32_f16(false, a, false, b,
                                                            (short)0, acc[t], false, false);
        }
        if (which == 0) {
            v16h a2 = load_frag(arow2, k0, hf);
            #pragma unroll
            for (int t = 0; t < 4; ++t) {
                v16h b2 = load_frag(brow20 + t * 16 * HDIM, k0, hf);
                acc[t] = __builtin_amdgcn_wmma_f32_16x16x32_f16(false, a2, false, b2,
                                                                (short)0, acc[t], false, false);
            }
        }
    }

    if (which == 0) {
        #pragma unroll
        for (int t = 0; t < 4; ++t) {
            float bias = b_n[tn + t * 16 + r];
            #pragma unroll
            for (int v = 0; v < 8; ++v) {              // C/D: VGPR v -> row v + 8*half
                int m = v + 8 * hf;
                Out[(tm + m) * HDIM + tn + t * 16 + r] = tanhf(acc[t][v] + bias);
            }
        }
    } else {
        #pragma unroll
        for (int t = 0; t < 4; ++t) {
            #pragma unroll
            for (int v = 0; v < 8; ++v) {
                int m = v + 8 * hf;
                Out[(tm + m) * HDIM + tn + t * 16 + r] = acc[t][v];
            }
        }
    }
}

// hg[b,h] = max over l of G[(l*B+b), h]
__global__ __launch_bounds__(256) void hg_max_kernel(const float* __restrict__ G,
                                                     float* __restrict__ hg)
{
    int idx = blockIdx.x * blockDim.x + threadIdx.x;   // b*H + h
    if (idx >= BDIM * HDIM) return;
    int b = idx / HDIM, h = idx % HDIM;
    float m = -__builtin_inff();
    for (int l = 0; l < LDIM; ++l)
        m = fmaxf(m, G[(l * BDIM + b) * HDIM + h]);
    hg[idx] = m;
}

// C[b,h] = b_hid[h] + sum_k hg[b,k] * W3[k,h]   (W3 = W_hid[2H:])
__global__ __launch_bounds__(256) void hg3_kernel(const float* __restrict__ hg,
                                                  const float* __restrict__ W_hid,
                                                  const float* __restrict__ b_hid,
                                                  float* __restrict__ C)
{
    int idx = blockIdx.x * blockDim.x + threadIdx.x;   // b*H + h
    if (idx >= BDIM * HDIM) return;
    int b = idx / HDIM, h = idx % HDIM;
    const float* __restrict__ W3 = W_hid + 2 * HDIM * HDIM;
    float s = b_hid[h];
    for (int k = 0; k < HDIM; ++k)
        s += hg[b * HDIM + k] * W3[k * HDIM + h];
    C[idx] = s;
}

// One wave32 per (i,j,b) row: build the 768-vector, layernorm (wave reduction),
// elu, 768x7 dot with W_tag, sigmoid, triangular+mask scale. The 226 MB
// (L,L,B,H) intermediate of the reference is never materialized.
__global__ __launch_bounds__(256) void fuse_kernel(
    const float* __restrict__ ST, const float* __restrict__ EN,
    const float* __restrict__ C,
    const float* __restrict__ ln_g, const float* __restrict__ ln_b,
    const float* __restrict__ W_tag, const float* __restrict__ b_tag,
    const float* __restrict__ mask, float* __restrict__ out)
{
    const int wavesPerBlock = 256 / 32;
    int wave = blockIdx.x * wavesPerBlock + (threadIdx.x >> 5);
    int lane = threadIdx.x & 31;
    if (wave >= LDIM * LDIM * BDIM) return;

    int b  = wave % BDIM;
    int ij = wave / BDIM;
    int j  = ij % LDIM;
    int i  = ij / LDIM;

    float mscale = (j >= i) ? mask[i * BDIM + b] * mask[j * BDIM + b] : 0.0f;

    const float* __restrict__ st = ST + (i * BDIM + b) * HDIM;
    const float* __restrict__ en = EN + (j * BDIM + b) * HDIM;
    const float* __restrict__ cc = C + b * HDIM;

    const int PER = HDIM / 32;     // 24 elements per lane
    float v[PER];
    float s1 = 0.f, s2 = 0.f;
    #pragma unroll
    for (int t = 0; t < PER; ++t) {
        int h = lane + 32 * t;
        float x = st[h] + en[h] + cc[h];
        v[t] = x;
        s1 += x;
        s2 += x * x;
    }
    #pragma unroll
    for (int o = 16; o > 0; o >>= 1) {
        s1 += __shfl_xor(s1, o, 32);
        s2 += __shfl_xor(s2, o, 32);
    }
    const float invH = 1.0f / (float)HDIM;
    float mean = s1 * invH;
    float var  = s2 * invH - mean * mean;
    float rstd = rsqrtf(var + 1e-5f);

    float acc0 = 0.f, acc1 = 0.f, acc2 = 0.f, acc3 = 0.f, acc4 = 0.f, acc5 = 0.f, acc6 = 0.f;
    #pragma unroll
    for (int t = 0; t < PER; ++t) {
        int h = lane + 32 * t;
        float x = (v[t] - mean) * rstd * ln_g[h] + ln_b[h];
        x = (x > 0.f) ? x : expm1f(x);                // elu, alpha=1
        const float* __restrict__ w = W_tag + h * TDIM;
        acc0 += x * w[0]; acc1 += x * w[1]; acc2 += x * w[2]; acc3 += x * w[3];
        acc4 += x * w[4]; acc5 += x * w[5]; acc6 += x * w[6];
    }
    #pragma unroll
    for (int o = 16; o > 0; o >>= 1) {
        acc0 += __shfl_xor(acc0, o, 32);
        acc1 += __shfl_xor(acc1, o, 32);
        acc2 += __shfl_xor(acc2, o, 32);
        acc3 += __shfl_xor(acc3, o, 32);
        acc4 += __shfl_xor(acc4, o, 32);
        acc5 += __shfl_xor(acc5, o, 32);
        acc6 += __shfl_xor(acc6, o, 32);
    }
    if (lane == 0) {
        float a[TDIM] = {acc0, acc1, acc2, acc3, acc4, acc5, acc6};
        float* o = out + (size_t)wave * TDIM;
        #pragma unroll
        for (int t = 0; t < TDIM; ++t) {
            float s = 1.0f / (1.0f + expf(-(a[t] + b_tag[t])));
            o[t] = s * mscale;
        }
    }
}

extern "C" void kernel_launch(void* const* d_in, const int* in_sizes, int n_in,
                              void* d_out, int out_size, void* d_ws, size_t ws_size,
                              hipStream_t stream) {
    const float* h_ner   = (const float*)d_in[0];
    const float* h_share = (const float*)d_in[1];
    const float* mask    = (const float*)d_in[2];
    const float* W_n     = (const float*)d_in[3];
    const float* b_n     = (const float*)d_in[4];
    const float* W_hid   = (const float*)d_in[5];
    const float* b_hid   = (const float*)d_in[6];
    const float* ln_g    = (const float*)d_in[7];
    const float* ln_b    = (const float*)d_in[8];
    const float* W_tag   = (const float*)d_in[9];
    const float* b_tag   = (const float*)d_in[10];
    float* out = (float*)d_out;

    const int MAT = HDIM * HDIM;           // 589824

    // fp32 scratch
    float* fws = (float*)d_ws;
    float* G   = fws;                      // MAT
    float* STm = G + MAT;                  // MAT
    float* ENm = STm + MAT;                // MAT
    float* hg  = ENm + MAT;                // B*H
    float* C   = hg + BDIM * HDIM;         // B*H
    // f16 scratch (16-byte aligned: offset is a multiple of 16)
    _Float16* hws  = (_Float16*)(C + BDIM * HDIM);
    _Float16* Xs16 = hws;                  // MAT halves
    _Float16* Xn16 = Xs16 + MAT;
    _Float16* Wn1t = Xn16 + MAT;
    _Float16* Wn2t = Wn1t + MAT;
    _Float16* W1t  = Wn2t + MAT;
    _Float16* W2t  = W1t + MAT;
    // total workspace ~14.2 MB

    // prep: convert activations, transpose+convert weight planes
    cvt_f16_kernel<<<(MAT + 255) / 256, 256, 0, stream>>>(h_share, Xs16, MAT);
    cvt_f16_kernel<<<(MAT + 255) / 256, 256, 0, stream>>>(h_ner,   Xn16, MAT);
    dim3 tGrid(HDIM / 32, HDIM / 32);
    transpose_f16_kernel<<<tGrid, 256, 0, stream>>>(W_n,            Wn1t);
    transpose_f16_kernel<<<tGrid, 256, 0, stream>>>(W_n + MAT,      Wn2t);
    transpose_f16_kernel<<<tGrid, 256, 0, stream>>>(W_hid,          W1t);
    transpose_f16_kernel<<<tGrid, 256, 0, stream>>>(W_hid + MAT,    W2t);

    // main WMMA GEMMs: 48 x 12 tiles of 16x64, z = {G, ST, EN}
    dim3 gGrid(HDIM / 16, HDIM / 64, 3);
    gemm768_wmma_kernel<<<gGrid, dim3(32), 0, stream>>>(
        Xs16, Xn16, Wn1t, Wn2t, W1t, W2t, b_n, G, STm, ENm);

    int n = BDIM * HDIM;
    hg_max_kernel<<<(n + 255) / 256, 256, 0, stream>>>(G, hg);
    hg3_kernel<<<(n + 255) / 256, 256, 0, stream>>>(hg, W_hid, b_hid, C);

    int rows = LDIM * LDIM * BDIM;         // 73728 waves
    fuse_kernel<<<rows / 8, 256, 0, stream>>>(
        STm, ENm, C, ln_g, ln_b, W_tag, b_tag, mask, out);
}
